// HeteroGNN_83056077570927
// MI455X (gfx1250) — matrix operand order repaired and checked by reading, not compiled, not run on hardware
//
#include <hip/hip_runtime.h>

// ---------------------------------------------------------------------------
// HeteroGNN (3 node types, 2x GATv2 + BN + leakyrelu + linear classifier)
// CDNA5 / gfx1250:
//  - dense GEMMs: v_wmma_f32_16x16x32_bf16 (wave32 WMMA). 32xK A tile staged
//    into LDS with global_load_async_to_lds_b128 (+ s_wait_asynccnt), K loop
//    fully unrolled (K templated: 128/256) so the scheduler overlaps global
//    B loads with matrix ops; each B fragment feeds two WMMAs (32-row tile).
//  - softmax/BN/classifier in fp32 VALU; segment-max via ordered-uint atomicMax.
// ---------------------------------------------------------------------------

typedef __attribute__((ext_vector_type(16))) __bf16 v16bf;
typedef __attribute__((ext_vector_type(8)))  float  v8f;

union ABfrag { v16bf v; uint4 u[2]; };

__device__ __forceinline__ unsigned short f2bf(float f) {
    unsigned u = __float_as_uint(f);
    unsigned r = ((u >> 16) & 1u) + 0x7FFFu;          // round-to-nearest-even
    return (unsigned short)((u + r) >> 16);
}

// order-preserving float <-> uint encoding (for atomicMax on floats)
__device__ __forceinline__ unsigned f2ord(float f) {
    unsigned u = __float_as_uint(f);
    return (u & 0x80000000u) ? ~u : (u | 0x80000000u);
}
__device__ __forceinline__ float ord2f(unsigned e) {
    unsigned u = (e & 0x80000000u) ? (e ^ 0x80000000u) : ~e;
    return __uint_as_float(u);
}

// ---------------------------------------------------------------------------
// f32 -> bf16 elementwise convert
__global__ __launch_bounds__(256)
void k_f32_to_bf16(const float* __restrict__ X, unsigned short* __restrict__ Y, long n) {
    long i = (long)blockIdx.x * 256 + threadIdx.x;
    if (i < n) Y[i] = f2bf(X[i]);
}

// W[K,256] f32 (row-major, x@W) -> Wt[256,K] bf16 (row = output channel)
__global__ __launch_bounds__(256)
void k_w_transpose_bf16(const float* __restrict__ W, unsigned short* __restrict__ Wt, int K) {
    int idx = blockIdx.x * 256 + threadIdx.x;
    if (idx < K * 256) {
        int k = idx >> 8;        // idx / 256
        int d = idx & 255;       // idx % 256
        Wt[(size_t)d * K + k] = f2bf(W[idx]);
    }
}

// ---------------------------------------------------------------------------
// C[N,256] = A[N,K](bf16) * Bt[256,K]^T(bf16), f32 accumulate.  K in {128,256}.
// Block = 256 threads = 8 waves; block owns 32 rows, wave w owns cols
// [32w, 32w+32) -> each wave produces a 32x32 patch (4 accumulators), and each
// B fragment is consumed by two WMMAs.
// Fragment layouts follow CDNA5 ISA 7.12.2 (wave32):
//   A 16x32 bf16: lanes 0-15 = rows, K {0..7,16..23}; lanes 16-31 same rows,
//                 K {8..15,24..31}  -> two 16B reads per lane (from LDS).
//   B 32x16 bf16: lane&15 = column, K {8h..8h+7, 16+8h..16+8h+7} -> two 16B reads.
//   C/D f32: element (M = r + 8*half, Ncol = lane&15) in acc[r].
template <int K>
__global__ __launch_bounds__(256)
void k_gemm_wmma(const unsigned short* __restrict__ A,
                 const unsigned short* __restrict__ Bt,
                 float* __restrict__ C, int N) {
    __shared__ __align__(16) unsigned short sA[32 * K];   // 8KB / 16KB

    const int tid  = threadIdx.x;
    const int lane = tid & 31;
    const int wave = tid >> 5;                // 0..7
    const int row0 = blockIdx.x * 32;
    const int col0 = wave * 32;
    const int l15  = lane & 15;
    const int half = lane >> 4;               // 0 or 1

    // ---- stage A tile (32 rows x K) into LDS: async DMA, 16B chunks --------
    {
        constexpr int cpr     = K / 8;        // 16B chunks per row
        constexpr int nchunks = 32 * cpr;     // 512 (K=128) / 1024 (K=256)
        for (int c = tid; c < nchunks; c += 256) {
            int r  = c / cpr;
            int kc = (c % cpr) * 8;
            int gr = row0 + r;
            if (gr >= N) gr = N - 1;          // clamp padded rows (discarded)
            unsigned long long gaddr =
                (unsigned long long)(const void*)(A + (size_t)gr * K + kc);
            unsigned laddr = (unsigned)(unsigned long long)(const void*)&sA[r * K + kc];
            asm volatile("global_load_async_to_lds_b128 %0, %1, off"
                         :: "v"(laddr), "v"(gaddr) : "memory");
        }
        asm volatile("s_wait_asynccnt 0x0" ::: "memory");
        __syncthreads();
    }

    v8f acc00 = {0.f, 0.f, 0.f, 0.f, 0.f, 0.f, 0.f, 0.f};  // rows 0-15 , cols +0
    v8f acc01 = {0.f, 0.f, 0.f, 0.f, 0.f, 0.f, 0.f, 0.f};  // rows 0-15 , cols +16
    v8f acc10 = {0.f, 0.f, 0.f, 0.f, 0.f, 0.f, 0.f, 0.f};  // rows 16-31, cols +0
    v8f acc11 = {0.f, 0.f, 0.f, 0.f, 0.f, 0.f, 0.f, 0.f};  // rows 16-31, cols +16

    const unsigned short* bBase = Bt + (size_t)(col0 + l15) * K + half * 8;
    constexpr size_t bS16 = (size_t)16 * K;   // 16 columns ahead

#pragma unroll
    for (int k0 = 0; k0 < K; k0 += 32) {
        ABfrag a0, a1, b0, b1;
        const unsigned short* lp0 = &sA[l15 * K + k0 + half * 8];
        a0.u[0] = *(const uint4*)(lp0);
        a0.u[1] = *(const uint4*)(lp0 + 16);
        const unsigned short* lp1 = lp0 + 16 * K;
        a1.u[0] = *(const uint4*)(lp1);
        a1.u[1] = *(const uint4*)(lp1 + 16);

        const unsigned short* bp = bBase + k0;
        b0.u[0] = *(const uint4*)(bp);
        b0.u[1] = *(const uint4*)(bp + 16);
        b1.u[0] = *(const uint4*)(bp + bS16);
        b1.u[1] = *(const uint4*)(bp + bS16 + 16);

        acc00 = __builtin_amdgcn_wmma_f32_16x16x32_bf16(false, a0.v, false, b0.v,
                                                        (short)0, acc00, false, false);
        acc01 = __builtin_amdgcn_wmma_f32_16x16x32_bf16(false, a0.v, false, b1.v,
                                                        (short)0, acc01, false, false);
        acc10 = __builtin_amdgcn_wmma_f32_16x16x32_bf16(false, a1.v, false, b0.v,
                                                        (short)0, acc10, false, false);
        acc11 = __builtin_amdgcn_wmma_f32_16x16x32_bf16(false, a1.v, false, b1.v,
                                                        (short)0, acc11, false, false);
    }

#pragma unroll
    for (int r = 0; r < 8; ++r) {
        int m0 = row0 + r + 8 * half;
        if (m0 < N) {
            C[(size_t)m0 * 256 + col0 + l15]      = acc00[r];
            C[(size_t)m0 * 256 + col0 + 16 + l15] = acc01[r];
        }
        int m1 = m0 + 16;
        if (m1 < N) {
            C[(size_t)m1 * 256 + col0 + l15]      = acc10[r];
            C[(size_t)m1 * 256 + col0 + 16 + l15] = acc11[r];
        }
    }
}

// ---------------------------------------------------------------------------
// One wave per edge (incl. self loops at e>=E). Lane l covers channels
// [8l, 8l+8) -> lanes 8h..8h+7 all belong to head h (HID=64).
// score[e][h] = sum_c att[h][c] * leakyrelu_0.2(hl[s][h,c] + hr[d][h,c])
__global__ __launch_bounds__(256)
void k_edge_score(const int* __restrict__ src, const int* __restrict__ dst,
                  int E, int N,
                  const float* __restrict__ hl, const float* __restrict__ hr,
                  const float* __restrict__ att,            // [4*64]
                  float* __restrict__ score,                // [Etot*4]
                  unsigned* __restrict__ mxbits) {          // [N*4]
    int e = blockIdx.x * 8 + (threadIdx.x >> 5);
    int lane = threadIdx.x & 31;
    int Etot = E + N;
    if (e >= Etot) return;
    int s = (e < E) ? src[e] : (e - E);
    int d = (e < E) ? dst[e] : (e - E);

    const float4* pl = (const float4*)(hl + (size_t)s * 256) + lane * 2;
    const float4* pr = (const float4*)(hr + (size_t)d * 256) + lane * 2;
    const float4* pa = (const float4*)(att) + lane * 2;

    float part = 0.f;
#pragma unroll
    for (int j = 0; j < 2; ++j) {
        float4 l = pl[j], r = pr[j], a = pa[j];
        float v;
        v = l.x + r.x; v = v > 0.f ? v : 0.2f * v; part += a.x * v;
        v = l.y + r.y; v = v > 0.f ? v : 0.2f * v; part += a.y * v;
        v = l.z + r.z; v = v > 0.f ? v : 0.2f * v; part += a.z * v;
        v = l.w + r.w; v = v > 0.f ? v : 0.2f * v; part += a.w * v;
    }
    // reduce groups of 8 lanes (one head each)
    part += __shfl_xor(part, 1, 32);
    part += __shfl_xor(part, 2, 32);
    part += __shfl_xor(part, 4, 32);
    if ((lane & 7) == 0) {
        int h = lane >> 3;
        score[(size_t)e * 4 + h] = part;
        atomicMax(&mxbits[(size_t)d * 4 + h], f2ord(part));
    }
}

// ex = exp(score - mx[d]); den[d] += ex; score slot overwritten with ex.
__global__ __launch_bounds__(256)
void k_edge_exp(const int* __restrict__ dst, int E, int N,
                float* __restrict__ score,
                const unsigned* __restrict__ mxbits,
                float* __restrict__ den) {
    int idx = blockIdx.x * 256 + threadIdx.x;      // over Etot*4
    int Etot = E + N;
    if (idx >= Etot * 4) return;
    int e = idx >> 2, h = idx & 3;
    int d = (e < E) ? dst[e] : (e - E);
    float m = ord2f(mxbits[(size_t)d * 4 + h]);
    float ex = expf(score[idx] - m);
    score[idx] = ex;
    atomicAdd(&den[(size_t)d * 4 + h], ex);
}

// out[d][ch] += (ex/den) * hl[s][ch]; one wave per edge, 8 ch per lane.
__global__ __launch_bounds__(256)
void k_edge_aggr(const int* __restrict__ src, const int* __restrict__ dst,
                 int E, int N,
                 const float* __restrict__ hl, const float* __restrict__ ex,
                 const float* __restrict__ den, float* __restrict__ out) {
    int e = blockIdx.x * 8 + (threadIdx.x >> 5);
    int lane = threadIdx.x & 31;
    int Etot = E + N;
    if (e >= Etot) return;
    int s = (e < E) ? src[e] : (e - E);
    int d = (e < E) ? dst[e] : (e - E);
    int h = lane >> 3;
    float alpha = ex[(size_t)e * 4 + h] / (den[(size_t)d * 4 + h] + 1e-16f);

    const float4* pl = (const float4*)(hl + (size_t)s * 256) + lane * 2;
    float* po = out + (size_t)d * 256 + lane * 8;
    float4 l0 = pl[0], l1 = pl[1];
    atomicAdd(po + 0, alpha * l0.x);
    atomicAdd(po + 1, alpha * l0.y);
    atomicAdd(po + 2, alpha * l0.z);
    atomicAdd(po + 3, alpha * l0.w);
    atomicAdd(po + 4, alpha * l1.x);
    atomicAdd(po + 5, alpha * l1.y);
    atomicAdd(po + 6, alpha * l1.z);
    atomicAdd(po + 7, alpha * l1.w);
}

// h += bias; accumulate per-channel sum / sumsq (stats[0..255]=sum, [256..511]=sumsq)
__global__ __launch_bounds__(256)
void k_bias_stats(float* __restrict__ h, const float* __restrict__ bias,
                  int N, float* __restrict__ stats) {
    int ch = threadIdx.x;                // 0..255
    int r0 = blockIdx.x * 64;
    int r1 = r0 + 64; if (r1 > N) r1 = N;
    float b = bias[ch];
    float s = 0.f, s2 = 0.f;
    for (int r = r0; r < r1; ++r) {
        float v = h[(size_t)r * 256 + ch] + b;
        h[(size_t)r * 256 + ch] = v;
        s += v; s2 += v * v;
    }
    atomicAdd(&stats[ch], s);
    atomicAdd(&stats[256 + ch], s2);
}

// batchnorm + leakyrelu(0.01); writes f32 in place and bf16 copy (next GEMM A)
__global__ __launch_bounds__(256)
void k_bn_act(float* __restrict__ h, unsigned short* __restrict__ hbf,
              const float* __restrict__ stats,
              const float* __restrict__ g, const float* __restrict__ b, int N) {
    int idx = blockIdx.x * 256 + threadIdx.x;
    if (idx >= N * 256) return;
    int ch = idx & 255;
    float invN = 1.f / (float)N;
    float mu = stats[ch] * invN;
    float var = stats[256 + ch] * invN - mu * mu;
    float v = (h[idx] - mu) * rsqrtf(var + 1e-5f) * g[ch] + b[ch];
    v = v > 0.f ? v : 0.01f * v;
    h[idx] = v;
    hbf[idx] = f2bf(v);
}

// out[n][0..1] = h[n] . cW + cb  (one wave per node, shuffle reduce)
__global__ __launch_bounds__(256)
void k_classifier(const float* __restrict__ h, const float* __restrict__ cW,
                  const float* __restrict__ cb, float* __restrict__ out, int N) {
    int n = blockIdx.x * 8 + (threadIdx.x >> 5);
    int lane = threadIdx.x & 31;
    if (n >= N) return;
    const float* hp = h + (size_t)n * 256 + lane * 8;
    float a0 = 0.f, a1 = 0.f;
#pragma unroll
    for (int j = 0; j < 8; ++j) {
        float v = hp[j];
        int ch = lane * 8 + j;
        a0 += v * cW[ch * 2 + 0];
        a1 += v * cW[ch * 2 + 1];
    }
#pragma unroll
    for (int m = 16; m >= 1; m >>= 1) {
        a0 += __shfl_xor(a0, m, 32);
        a1 += __shfl_xor(a1, m, 32);
    }
    if (lane == 0) {
        out[(size_t)n * 2 + 0] = a0 + cb[0];
        out[(size_t)n * 2 + 1] = a1 + cb[1];
    }
}

// ---------------------------------------------------------------------------
extern "C" void kernel_launch(void* const* d_in, const int* in_sizes, int n_in,
                              void* d_out, int out_size, void* d_ws, size_t ws_size,
                              hipStream_t stream) {
    (void)in_sizes; (void)n_in; (void)out_size; (void)ws_size;

    const float* x[3]  = {(const float*)d_in[0], (const float*)d_in[1], (const float*)d_in[2]};
    const int*   ei[3] = {(const int*)d_in[3], (const int*)d_in[4], (const int*)d_in[5]};
    const float* Wl1  = (const float*)d_in[6];
    const float* Wr1  = (const float*)d_in[7];
    const float* att1 = (const float*)d_in[8];
    const float* b1   = (const float*)d_in[9];
    const float* Wl2  = (const float*)d_in[10];
    const float* Wr2  = (const float*)d_in[11];
    const float* att2 = (const float*)d_in[12];
    const float* b2   = (const float*)d_in[13];
    const float* bng  = (const float*)d_in[14];
    const float* bnb  = (const float*)d_in[15];
    const float* cW   = (const float*)d_in[16];
    const float* cb   = (const float*)d_in[17];

    const int Ns[3] = {20000, 50000, 30000};
    const int Es[3] = {160000, 400000, 240000};

    // workspace carve (sized for max N=50000, max Etot=450000) ~190 MB
    char* ws = (char*)d_ws;
    size_t off = 0;
    auto carve = [&](size_t bytes) -> char* {
        char* p = ws + off;
        off += (bytes + 255) & ~(size_t)255;
        return p;
    };
    unsigned short* XBF = (unsigned short*)carve(50000ull * 256 * 2);
    unsigned short* WTl = (unsigned short*)carve(256ull * 256 * 2);
    unsigned short* WTr = (unsigned short*)carve(256ull * 256 * 2);
    float*    HL  = (float*)carve(50000ull * 256 * 4);
    float*    HR  = (float*)carve(50000ull * 256 * 4);
    float*    HO  = (float*)carve(50000ull * 256 * 4);
    float*    SC  = (float*)carve(450000ull * 4 * 4);
    unsigned* MX  = (unsigned*)carve(50000ull * 4 * 4);
    float*    DEN = (float*)carve(50000ull * 4 * 4);
    float*    ST  = (float*)carve(512 * 4);

    float* out = (float*)d_out;
    size_t outOff = 0;

    for (int i = 0; i < 3; ++i) {
        const int N = Ns[i], E = Es[i], Etot = E + N;
        const int* srcp = ei[i];
        const int* dstp = ei[i] + E;

        {   // x -> bf16 (N x 128)
            long n = (long)N * 128;
            k_f32_to_bf16<<<(int)((n + 255) / 256), 256, 0, stream>>>(x[i], XBF, n);
        }

        for (int layer = 0; layer < 2; ++layer) {
            const int K = (layer == 0) ? 128 : 256;
            const float* Wl  = (layer == 0) ? Wl1 + (size_t)i * 128 * 256
                                            : Wl2 + (size_t)i * 256 * 256;
            const float* Wr  = (layer == 0) ? Wr1 + (size_t)i * 128 * 256
                                            : Wr2 + (size_t)i * 256 * 256;
            const float* att = ((layer == 0) ? att1 : att2) + (size_t)i * 256;
            const float* bia = ((layer == 0) ? b1 : b2) + (size_t)i * 256;

            k_w_transpose_bf16<<<(K * 256 + 255) / 256, 256, 0, stream>>>(Wl, WTl, K);
            k_w_transpose_bf16<<<(K * 256 + 255) / 256, 256, 0, stream>>>(Wr, WTr, K);

            const int gblk = (N + 31) / 32;
            if (K == 128) {
                k_gemm_wmma<128><<<gblk, 256, 0, stream>>>(XBF, WTl, HL, N);
                k_gemm_wmma<128><<<gblk, 256, 0, stream>>>(XBF, WTr, HR, N);
            } else {
                k_gemm_wmma<256><<<gblk, 256, 0, stream>>>(XBF, WTl, HL, N);
                k_gemm_wmma<256><<<gblk, 256, 0, stream>>>(XBF, WTr, HR, N);
            }

            hipMemsetAsync(MX,  0, (size_t)N * 4 * 4,   stream);
            hipMemsetAsync(DEN, 0, (size_t)N * 4 * 4,   stream);
            hipMemsetAsync(HO,  0, (size_t)N * 256 * 4, stream);
            hipMemsetAsync(ST,  0, 512 * 4,             stream);

            k_edge_score<<<(Etot + 7) / 8, 256, 0, stream>>>(srcp, dstp, E, N, HL, HR, att, SC, MX);
            k_edge_exp<<<(Etot * 4 + 255) / 256, 256, 0, stream>>>(dstp, E, N, SC, MX, DEN);
            k_edge_aggr<<<(Etot + 7) / 8, 256, 0, stream>>>(srcp, dstp, E, N, HL, SC, DEN, HO);

            k_bias_stats<<<(N + 63) / 64, 256, 0, stream>>>(HO, bia, N, ST);
            k_bn_act<<<(N * 256 + 255) / 256, 256, 0, stream>>>(
                HO, XBF, ST, bng + (size_t)i * 256, bnb + (size_t)i * 256, N);
        }

        k_classifier<<<(N + 7) / 8, 256, 0, stream>>>(HO, cW, cb, out + outOff, N);
        outOff += (size_t)N * 2;
    }
}
